// GNNBlock_26603027432072
// MI455X (gfx1250) — matrix operand order repaired and checked by reading, not compiled, not run on hardware
//
#include <hip/hip_runtime.h>
#include <hip/hip_bf16.h>
#include <math.h>

#define N_NODES 50000
#define N_EDGES 800000
#define N_MSG   (N_EDGES + N_NODES)
#define F       128
#define HEADS   4
#define CPH     32
#define NEG_SLOPE 0.2f
#define BN_EPS  1e-5f

typedef __attribute__((ext_vector_type(16))) __bf16 v16bf;
typedef __attribute__((ext_vector_type(8)))  float  v8f;

// ---------------------------------------------------------------------------
// monotonic uint encoding for float atomicMax (order-preserving, deterministic)
// ---------------------------------------------------------------------------
__device__ __forceinline__ unsigned enc_f32(float f) {
  unsigned u = __float_as_uint(f);
  return (u & 0x80000000u) ? ~u : (u | 0x80000000u);
}
__device__ __forceinline__ float dec_f32(unsigned u) {
  u = (u & 0x80000000u) ? (u & 0x7FFFFFFFu) : ~u;
  return __uint_as_float(u);
}

__device__ __forceinline__ void msg_src_dst(const int* __restrict__ ei, int m,
                                            int& s, int& d) {
  if (m < N_EDGES) { s = ei[m]; d = ei[N_EDGES + m]; }
  else             { s = m - N_EDGES; d = s; }
}

__device__ __forceinline__ float leaky(float e) {
  return e > 0.0f ? e : NEG_SLOPE * e;
}

// split one float4 into bf16 hi/lo halves at vector element offset `base`
__device__ __forceinline__ void split4(float4 f, v16bf& hi, v16bf& lo, int base) {
  float vv[4] = {f.x, f.y, f.z, f.w};
#pragma unroll
  for (int i = 0; i < 4; ++i) {
    __bf16 hb = (__bf16)vv[i];
    hi[base + i] = hb;
    lo[base + i] = (__bf16)(vv[i] - (float)hb);
  }
}

// ---------------------------------------------------------------------------
// K1: init output to bias, zero softmax/statistics scratch
// ---------------------------------------------------------------------------
__global__ void init_ws(float* __restrict__ out, const float* __restrict__ bias,
                        unsigned* __restrict__ emax_u, float* __restrict__ denom,
                        float* __restrict__ fsum, float* __restrict__ fsumsq) {
  size_t idx = (size_t)blockIdx.x * blockDim.x + threadIdx.x;
  if (idx < (size_t)N_NODES * F) out[idx] = bias[idx & (F - 1)];
  if (idx < (size_t)N_NODES * HEADS) { emax_u[idx] = 0u; denom[idx] = 0.0f; }
  if (idx < F) { fsum[idx] = 0.0f; fsumsq[idx] = 0.0f; }
}

// ---------------------------------------------------------------------------
// K2: h = x @ W^T via v_wmma_f32_16x16x32_bf16, split-bf16 (hi/lo) for ~fp32
// accuracy.  Block = 256 threads = 8 waves; wave w owns N-tile w (0..7),
// block b owns M-tile b (N=50000 = 3125*16 exactly).
//
// Contiguous K-groups per lane let us fetch with global_load_b128:
//   A: elements 0..7  <- x[m][kt + half*8      + 0..7]
//      elements 8..15 <- x[m][kt + 16 + half*8 + 0..7]
//   B: elements 0..7  <- w[n][kt + half*8      + 0..7]
//      elements 8..15 <- w[n][kt + 16 + half*8 + 0..7]
// ---------------------------------------------------------------------------
__global__ void __launch_bounds__(256)
gat_gemm_wmma(const float* __restrict__ x, const float* __restrict__ w,
              float* __restrict__ h) {
  const int lane = threadIdx.x & 31;
  const int wave = threadIdx.x >> 5;
  const int half = lane >> 4;         // which 16-lane half of the wave
  const int l16  = lane & 15;
  const int m_base = blockIdx.x * 16;
  const int n_base = wave * 16;

  const float* xrow = x + (size_t)(m_base + l16) * F;  // A row this lane feeds
  const float* wrow = w + (size_t)(n_base + l16) * F;  // B col (= W row)

  v8f c = {0.f, 0.f, 0.f, 0.f, 0.f, 0.f, 0.f, 0.f};

  for (int kt = 0; kt < F; kt += 32) {
    v16bf a_hi, a_lo, b_hi, b_lo;

    const int aBase0 = kt + half * 8;        // A K-group 0 (elements 0..7)
    const int aBase1 = kt + 16 + half * 8;   // A K-group 1 (elements 8..15)
    const int bBase0 = kt + half * 8;        // B K-group 0 (elements 0..7)
    const int bBase1 = kt + 16 + half * 8;   // B K-group 1 (elements 8..15)

    split4(*(const float4*)(xrow + aBase0),     a_hi, a_lo, 0);
    split4(*(const float4*)(xrow + aBase0 + 4), a_hi, a_lo, 4);
    split4(*(const float4*)(xrow + aBase1),     a_hi, a_lo, 8);
    split4(*(const float4*)(xrow + aBase1 + 4), a_hi, a_lo, 12);

    split4(*(const float4*)(wrow + bBase0),     b_hi, b_lo, 0);
    split4(*(const float4*)(wrow + bBase0 + 4), b_hi, b_lo, 4);
    split4(*(const float4*)(wrow + bBase1),     b_hi, b_lo, 8);
    split4(*(const float4*)(wrow + bBase1 + 4), b_hi, b_lo, 12);

    // (hi+lo)*(Hi+Lo) ~= hi*Hi + hi*Lo + lo*Hi   (drop lo*Lo, ~2^-16 rel err)
    c = __builtin_amdgcn_wmma_f32_16x16x32_bf16(false, a_hi, false, b_hi,
                                                (short)0, c, false, false);
    c = __builtin_amdgcn_wmma_f32_16x16x32_bf16(false, a_hi, false, b_lo,
                                                (short)0, c, false, false);
    c = __builtin_amdgcn_wmma_f32_16x16x32_bf16(false, a_lo, false, b_hi,
                                                (short)0, c, false, false);
  }

#pragma unroll
  for (int r = 0; r < 8; ++r) {
    int row = m_base + r + half * 8;           // C layout: M = r (+8 hi-half)
    h[(size_t)row * F + n_base + l16] = c[r];
  }
}

// ---------------------------------------------------------------------------
// K3: a_src[n,h] = <h[n,h,:], att_src[h,:]>,  a_dst likewise
// ---------------------------------------------------------------------------
__global__ void att_dots(const float* __restrict__ h,
                         const float* __restrict__ att_src,
                         const float* __restrict__ att_dst,
                         float* __restrict__ a_src, float* __restrict__ a_dst) {
  int idx = blockIdx.x * blockDim.x + threadIdx.x;   // over N*HEADS
  if (idx >= N_NODES * HEADS) return;
  int n = idx >> 2, hh = idx & 3;
  const float* hp = h + (size_t)n * F + hh * CPH;
  const float* as = att_src + hh * CPH;
  const float* ad = att_dst + hh * CPH;
  float s = 0.f, d = 0.f;
#pragma unroll
  for (int cI = 0; cI < CPH; ++cI) {
    float v = hp[cI];
    s += v * as[cI];
    d += v * ad[cI];
  }
  a_src[idx] = s; a_dst[idx] = d;
}

// ---------------------------------------------------------------------------
// K4: segment max of leaky-relu logits (monotonic-uint atomicMax)
// ---------------------------------------------------------------------------
__global__ void edge_max(const int* __restrict__ ei,
                         const float* __restrict__ a_src,
                         const float* __restrict__ a_dst,
                         unsigned* __restrict__ emax_u) {
  int idx = blockIdx.x * blockDim.x + threadIdx.x;   // over N_MSG*HEADS
  if (idx >= N_MSG * HEADS) return;
  int m = idx >> 2, hh = idx & 3;
  int s, d; msg_src_dst(ei, m, s, d);
  float e = leaky(a_src[s * HEADS + hh] + a_dst[d * HEADS + hh]);
  atomicMax(&emax_u[d * HEADS + hh], enc_f32(e));
}

// ---------------------------------------------------------------------------
// K5: segment sum of exp(e - emax[dst])
// ---------------------------------------------------------------------------
__global__ void edge_denom(const int* __restrict__ ei,
                           const float* __restrict__ a_src,
                           const float* __restrict__ a_dst,
                           const unsigned* __restrict__ emax_u,
                           float* __restrict__ denom) {
  int idx = blockIdx.x * blockDim.x + threadIdx.x;
  if (idx >= N_MSG * HEADS) return;
  int m = idx >> 2, hh = idx & 3;
  int s, d; msg_src_dst(ei, m, s, d);
  float e = leaky(a_src[s * HEADS + hh] + a_dst[d * HEADS + hh]);
  float ee = __expf(e - dec_f32(emax_u[d * HEADS + hh]));
  atomicAdd(&denom[d * HEADS + hh], ee);
}

// ---------------------------------------------------------------------------
// K6: out[dst] += alpha * h[src]  — one 128-thread block per message,
// fully coalesced 512B gather + 512B f32 atomic scatter.
// ---------------------------------------------------------------------------
__global__ void __launch_bounds__(128)
aggregate(const int* __restrict__ ei, const float* __restrict__ h,
          const float* __restrict__ a_src, const float* __restrict__ a_dst,
          const unsigned* __restrict__ emax_u, const float* __restrict__ denom,
          float* __restrict__ out) {
  int m = blockIdx.x;
  int f = threadIdx.x;
  int hh = f >> 5;
  int s, d; msg_src_dst(ei, m, s, d);
  float e = leaky(a_src[s * HEADS + hh] + a_dst[d * HEADS + hh]);
  float alpha = __expf(e - dec_f32(emax_u[d * HEADS + hh])) /
                (denom[d * HEADS + hh] + 1e-16f);
  atomicAdd(&out[(size_t)d * F + f], h[(size_t)s * F + f] * alpha);
}

// ---------------------------------------------------------------------------
// K7: per-feature sum / sumsq (coalesced: thread = feature, block = row chunk)
// ---------------------------------------------------------------------------
__global__ void __launch_bounds__(128)
bn_stats(const float* __restrict__ out, float* __restrict__ fsum,
         float* __restrict__ fsumsq) {
  int f = threadIdx.x;
  int row0 = blockIdx.x * 250;
  int rend = row0 + 250; if (rend > N_NODES) rend = N_NODES;
  float s = 0.f, q = 0.f;
  for (int r = row0; r < rend; ++r) {
    float v = out[(size_t)r * F + f];
    s += v; q += v * v;
  }
  atomicAdd(&fsum[f], s);
  atomicAdd(&fsumsq[f], q);
}

// ---------------------------------------------------------------------------
// K8: in-place batchnorm (biased var) + ReLU
// ---------------------------------------------------------------------------
__global__ void bn_apply(float* __restrict__ out, const float* __restrict__ fsum,
                         const float* __restrict__ fsumsq,
                         const float* __restrict__ gamma,
                         const float* __restrict__ beta) {
  size_t idx = (size_t)blockIdx.x * blockDim.x + threadIdx.x;
  if (idx >= (size_t)N_NODES * F) return;
  int f = (int)(idx & (F - 1));
  const float invN = 1.0f / (float)N_NODES;
  float mean = fsum[f] * invN;
  float var  = fsumsq[f] * invN - mean * mean;
  float v = out[idx];
  float y = (v - mean) * rsqrtf(var + BN_EPS) * gamma[f] + beta[f];
  out[idx] = y > 0.0f ? y : 0.0f;
}

// ---------------------------------------------------------------------------
extern "C" void kernel_launch(void* const* d_in, const int* in_sizes, int n_in,
                              void* d_out, int out_size, void* d_ws, size_t ws_size,
                              hipStream_t stream) {
  const float* x       = (const float*)d_in[0];   // [N, 128]
  const int*   ei      = (const int*)  d_in[1];   // [2, E] flat
  const float* lin_w   = (const float*)d_in[2];   // [128, 128]
  const float* att_src = (const float*)d_in[3];   // [4, 32]
  const float* att_dst = (const float*)d_in[4];   // [4, 32]
  const float* bias    = (const float*)d_in[5];   // [128]
  const float* gamma   = (const float*)d_in[6];   // [128]
  const float* beta    = (const float*)d_in[7];   // [128]
  float* out = (float*)d_out;                     // [N, 128]

  // workspace carve-up (~29 MB)
  float*    h      = (float*)d_ws;                       // N*F
  float*    a_src  = h + (size_t)N_NODES * F;            // N*H
  float*    a_dst  = a_src + (size_t)N_NODES * HEADS;    // N*H
  unsigned* emax_u = (unsigned*)(a_dst + (size_t)N_NODES * HEADS); // N*H
  float*    denom  = (float*)(emax_u + (size_t)N_NODES * HEADS);   // N*H
  float*    fsum   = denom + (size_t)N_NODES * HEADS;    // F
  float*    fsumsq = fsum + F;                           // F
  (void)in_sizes; (void)n_in; (void)out_size; (void)ws_size;

  const size_t NF = (size_t)N_NODES * F;

  init_ws<<<(unsigned)((NF + 255) / 256), 256, 0, stream>>>(
      out, bias, emax_u, denom, fsum, fsumsq);

  gat_gemm_wmma<<<N_NODES / 16, 256, 0, stream>>>(x, lin_w, h);

  att_dots<<<(N_NODES * HEADS + 255) / 256, 256, 0, stream>>>(
      h, att_src, att_dst, a_src, a_dst);

  edge_max<<<(N_MSG * HEADS + 255) / 256, 256, 0, stream>>>(
      ei, a_src, a_dst, emax_u);

  edge_denom<<<(N_MSG * HEADS + 255) / 256, 256, 0, stream>>>(
      ei, a_src, a_dst, emax_u, denom);

  aggregate<<<N_MSG, 128, 0, stream>>>(
      ei, h, a_src, a_dst, emax_u, denom, out);

  bn_stats<<<(N_NODES + 249) / 250, 128, 0, stream>>>(out, fsum, fsumsq);

  bn_apply<<<(unsigned)((NF + 255) / 256), 256, 0, stream>>>(
      out, fsum, fsumsq, gamma, beta);
}
